// ReduNetGCNLayer_85177791414410
// MI455X (gfx1250) — matrix operand order repaired and checked by reading, not compiled, not run on hardware
//
#include <hip/hip_runtime.h>
#include <hip/hip_bf16.h>

typedef __bf16 bf16_t;
typedef __attribute__((ext_vector_type(8)))  __bf16 v8bf;
typedef __attribute__((ext_vector_type(16))) __bf16 v16bf;
typedef __attribute__((ext_vector_type(8)))  float  v8f;

#define NN    100000
#define DD    256
#define KK    10
#define NEDGE 3200000
#define ETA   0.5f
#define ALPHA 0.5f
#define LNEPS 1e-5f

// ---------------- zero helper ----------------
__global__ void zero_kernel(float* __restrict__ p, size_t n) {
  size_t i = (size_t)blockIdx.x * blockDim.x + threadIdx.x;
  size_t stride = (size_t)gridDim.x * blockDim.x;
  for (; i < n; i += stride) p[i] = 0.0f;
}

// ---------------- gamma = softplus(log_gamma) ----------------
__global__ void gamma_kernel(const float* __restrict__ lg, float* __restrict__ g) {
  int t = threadIdx.x;
  if (t < KK) {
    float x = lg[t];
    g[t] = (x > 20.0f) ? x : log1pf(expf(x));
  }
}

// ---------------- SpMM: H_agg[rows[e]] += A_vals[e] * H[cols[e]] ----------------
// one wave (32 lanes) per edge, 8 floats per lane (whole 256-wide row per wave)
__global__ void spmm_kernel(const float* __restrict__ H, const float* __restrict__ Av,
                            const int* __restrict__ rows, const int* __restrict__ cols,
                            float* __restrict__ Hagg) {
  int wave = blockIdx.x * (blockDim.x >> 5) + (threadIdx.x >> 5);
  int lane = threadIdx.x & 31;
  if (wave >= NEDGE) return;
  int r = rows[wave], c = cols[wave];
  float a = Av[wave];
  const float4* src = (const float4*)(H + (size_t)c * DD + lane * 8);
  float4 x0 = src[0], x1 = src[1];
  float* dst = Hagg + (size_t)r * DD + lane * 8;
  atomicAdd(dst + 0, a * x0.x); atomicAdd(dst + 1, a * x0.y);
  atomicAdd(dst + 2, a * x0.z); atomicAdd(dst + 3, a * x0.w);
  atomicAdd(dst + 4, a * x1.x); atomicAdd(dst + 5, a * x1.y);
  atomicAdd(dst + 6, a * x1.z); atomicAdd(dst + 7, a * x1.w);
}

// ---------------- LayerNorm per row; write Hn (f32) and Hnb (bf16) ----------------
__global__ void ln_kernel(const float* __restrict__ Hagg, const float* __restrict__ w,
                          const float* __restrict__ b, float* __restrict__ Hn,
                          bf16_t* __restrict__ Hnb) {
  __shared__ float red[DD];
  int n = blockIdx.x, t = threadIdx.x;
  float x = Hagg[(size_t)n * DD + t];
  red[t] = x; __syncthreads();
  for (int s = 128; s > 0; s >>= 1) { if (t < s) red[t] += red[t + s]; __syncthreads(); }
  float mu = red[0] * (1.0f / DD);
  __syncthreads();
  float dx = x - mu;
  red[t] = dx * dx; __syncthreads();
  for (int s = 128; s > 0; s >>= 1) { if (t < s) red[t] += red[t + s]; __syncthreads(); }
  float var = red[0] * (1.0f / DD);
  float hn = dx * rsqrtf(var + LNEPS) * w[t] + b[t];
  Hn[(size_t)n * DD + t] = hn;
  Hnb[(size_t)n * DD + t] = (bf16_t)hn;
}

// ---------------- transpose Hnb (N x D, bf16) -> HnT (D x N, bf16) ----------------
__global__ void transpose_kernel(const unsigned short* __restrict__ src,
                                 unsigned short* __restrict__ dst) {
  __shared__ unsigned short tile[32][33];
  int n0 = blockIdx.x * 32, d0 = blockIdx.y * 32;
  int tx = threadIdx.x, ty = threadIdx.y;  // 32 x 8
  for (int i = 0; i < 4; i++) {
    int nr = ty + i * 8;
    tile[nr][tx] = src[(size_t)(n0 + nr) * DD + d0 + tx];
  }
  __syncthreads();
  for (int i = 0; i < 4; i++) {
    int dr = ty + i * 8;
    dst[(size_t)(d0 + dr) * NN + n0 + tx] = tile[tx][dr];
  }
}

// ---------------- transpose soft_labels -> Pt (K x N), plus column sums ----------------
__global__ void pt_kernel(const float* __restrict__ soft, float* __restrict__ Pt,
                          float* __restrict__ sums) {
  __shared__ float red[256];
  int t = threadIdx.x;
  int n = blockIdx.x * 256 + t;
  bool ok = (n < NN);
  float v[KK];
  for (int k = 0; k < KK; k++) {
    v[k] = ok ? soft[(size_t)n * KK + k] : 0.0f;
    if (ok) Pt[(size_t)k * NN + n] = v[k];
  }
  for (int k = 0; k < KK; k++) {
    red[t] = v[k]; __syncthreads();
    for (int s = 128; s > 0; s >>= 1) { if (t < s) red[t] += red[t + s]; __syncthreads(); }
    if (t == 0) atomicAdd(&sums[k], red[0]);
    __syncthreads();
  }
}

// ---------------- Gram: G_z[a][b] = sum_n w_z(n) Hn[n][a] Hn[n][b] ----------------
// z=0: w=1 ; z>=1: w = pi_{z-1}^2 applied entirely to the A side (f32, one rounding).
// One wave per (32x32 tile quad, z, N-split): 2x2 tiles -> 4 WMMAs per K-chunk,
// amortizing fragment loads and the weight scaling.
__global__ void gram_kernel(const bf16_t* __restrict__ HnT, const float* __restrict__ Pt,
                            float* __restrict__ G) {
  int pair = blockIdx.x;                  // 0..63  (Ip, Jp pairs of 16-tiles)
  int z = blockIdx.y;                     // 0..10
  int split = blockIdx.z;                 // 0..15
  int Ip = pair >> 3, Jp = pair & 7;
  int lane = threadIdx.x;
  int q = lane & 15, half = lane >> 4;
  int baseA = half ? 8 : 0;               // 16-bit A-fragment K pattern
  int baseB = half ? 8 : 0;               // 16-bit B-fragment K pattern
  const bf16_t* rowA0 = HnT + (size_t)(32 * Ip + q) * NN;
  const bf16_t* rowA1 = rowA0 + (size_t)16 * NN;
  const bf16_t* rowB0 = HnT + (size_t)(32 * Jp + q) * NN;
  const bf16_t* rowB1 = rowB0 + (size_t)16 * NN;
  const float* wz = (z > 0) ? (Pt + (size_t)(z - 1) * NN) : (const float*)0;
  v8f acc00 = {}, acc01 = {}, acc10 = {}, acc11 = {};
  for (int c = split; c < NN / 32; c += 16) {
    int n0 = c * 32;
    v8bf a0Lo = *(const v8bf*)(rowA0 + n0 + baseA);
    v8bf a0Hi = *(const v8bf*)(rowA0 + n0 + baseA + 16);
    v8bf a1Lo = *(const v8bf*)(rowA1 + n0 + baseA);
    v8bf a1Hi = *(const v8bf*)(rowA1 + n0 + baseA + 16);
    v8bf b0Lo = *(const v8bf*)(rowB0 + n0 + baseB);
    v8bf b0Hi = *(const v8bf*)(rowB0 + n0 + baseB + 16);
    v8bf b1Lo = *(const v8bf*)(rowB1 + n0 + baseB);
    v8bf b1Hi = *(const v8bf*)(rowB1 + n0 + baseB + 16);
    __builtin_prefetch(rowA0 + n0 + 512, 0, 1);
    __builtin_prefetch(rowB0 + n0 + 512, 0, 1);
    v16bf a0, a1, b0, b1;
    if (z == 0) {
      for (int j = 0; j < 8; j++) {
        a0[j] = a0Lo[j]; a0[j + 8] = a0Hi[j];
        a1[j] = a1Lo[j]; a1[j + 8] = a1Hi[j];
      }
    } else {
      v8f wl = *(const v8f*)(wz + n0 + baseA);
      v8f wh = *(const v8f*)(wz + n0 + baseA + 16);
      for (int j = 0; j < 8; j++) {
        float w2l = wl[j] * wl[j];
        float w2h = wh[j] * wh[j];
        a0[j]     = (bf16_t)((float)a0Lo[j] * w2l);
        a0[j + 8] = (bf16_t)((float)a0Hi[j] * w2h);
        a1[j]     = (bf16_t)((float)a1Lo[j] * w2l);
        a1[j + 8] = (bf16_t)((float)a1Hi[j] * w2h);
      }
    }
    for (int j = 0; j < 8; j++) {
      b0[j] = b0Lo[j]; b0[j + 8] = b0Hi[j];
      b1[j] = b1Lo[j]; b1[j + 8] = b1Hi[j];
    }
    acc00 = __builtin_amdgcn_wmma_f32_16x16x32_bf16(false, a0, false, b0, (short)0, acc00,
                                                    false, false);
    acc01 = __builtin_amdgcn_wmma_f32_16x16x32_bf16(false, a0, false, b1, (short)0, acc01,
                                                    false, false);
    acc10 = __builtin_amdgcn_wmma_f32_16x16x32_bf16(false, a1, false, b0, (short)0, acc10,
                                                    false, false);
    acc11 = __builtin_amdgcn_wmma_f32_16x16x32_bf16(false, a1, false, b1, (short)0, acc11,
                                                    false, false);
  }
  float* Gz = G + (size_t)z * 65536;
  int cb0 = 32 * Jp + q, cb1 = 32 * Jp + 16 + q;
  for (int v = 0; v < 8; v++) {
    int r0 = 32 * Ip + v + (half ? 8 : 0);
    int r1 = r0 + 16;
    atomicAdd(&Gz[r0 * 256 + cb0], acc00[v]);
    atomicAdd(&Gz[r0 * 256 + cb1], acc01[v]);
    atomicAdd(&Gz[r1 * 256 + cb0], acc10[v]);
    atomicAdd(&Gz[r1 * 256 + cb1], acc11[v]);
  }
}

// ---------------- In-place Gauss-Jordan inverse of M = I + s*G (SPD) ----------------
// one 1024-thread block per matrix; pivot row/col staged in LDS. Writes bf16 copy.
__global__ __launch_bounds__(1024) void inv_kernel(float* __restrict__ G,
                                                   const float* __restrict__ sums,
                                                   bf16_t* __restrict__ Eb) {
  __shared__ float prow[256];
  __shared__ float pcol[256];
  int z = blockIdx.x;
  float* M = G + (size_t)z * 65536;
  int tid = threadIdx.x;
  int c = tid & 255, g = tid >> 8;
  float s = (z == 0) ? (ALPHA / (float)NN) : (ALPHA / fmaxf(sums[z - 1], 1.0f));
  for (int r = g * 64; r < g * 64 + 64; r++) {
    float val = s * M[r * 256 + c];
    if (r == c) val += 1.0f;
    M[r * 256 + c] = val;
  }
  __syncthreads();
  for (int p = 0; p < 256; p++) {
    if (tid < 256) pcol[tid] = M[tid * 256 + p];
    __syncthreads();
    if (tid < 256) {
      float pd = 1.0f / pcol[p];
      float v = (tid == p) ? pd : M[p * 256 + tid] * pd;
      M[p * 256 + tid] = v;
      prow[tid] = v;
    }
    __syncthreads();
    for (int r = g * 64; r < g * 64 + 64; r++) {
      if (r == p) continue;
      float f = pcol[r];
      float cur = (c == p) ? 0.0f : M[r * 256 + c];
      M[r * 256 + c] = fmaf(-f, prow[c], cur);
    }
    __syncthreads();
  }
  for (int r = g * 64; r < g * 64 + 64; r++)
    Eb[(size_t)z * 65536 + r * 256 + c] = (bf16_t)M[r * 256 + c];
}

// ---------------- Apply: out = shrink(Hn + ETA*(Hn@E - sum_k g_k pi_k (.)@C_k)) ----
// 8 waves per block, same col-tile, 8 row-tiles. The 16x256 bf16 strip of E_z used
// for B fragments is staged once per z into LDS and shared by all 8 waves.
__global__ __launch_bounds__(256) void apply_kernel(
    const bf16_t* __restrict__ Hnb, const bf16_t* __restrict__ Eb,
    const float* __restrict__ Hn, const float* __restrict__ soft,
    const float* __restrict__ gamma, const float* __restrict__ thrp,
    float* __restrict__ out) {
  __shared__ __align__(16) bf16_t Bsh[16 * 256];
  int tid = threadIdx.x;
  int wave = tid >> 5, lane = tid & 31;
  int q = lane & 15, half = lane >> 4;
  int baseA = half ? 8 : 0, baseB = half ? 8 : 0, off8 = half ? 8 : 0;
  int rt = blockIdx.x * 8 + wave;
  bool active = rt < (NN / 16);
  int rowbase = rt * 16;
  int colbase = blockIdx.y * 16;
  int col = colbase + q;

  // preload A fragments (rows of Hn, bf16) for all 8 K-chunks
  v16bf A[8];
  if (active) {
    const bf16_t* arow = Hnb + (size_t)(rowbase + q) * DD;
    for (int kc = 0; kc < 8; kc++) {
      v8bf lo = *(const v8bf*)(arow + kc * 32 + baseA);
      v8bf hi = *(const v8bf*)(arow + kc * 32 + baseA + 16);
      for (int j = 0; j < 8; j++) { A[kc][j] = lo[j]; A[kc][j + 8] = hi[j]; }
    }
  }

  float total[8] = {0, 0, 0, 0, 0, 0, 0, 0};
  int srow = tid >> 4, schunk = tid & 15;  // cooperative staging coords
  for (int z = 0; z < 11; z++) {
    __syncthreads();
    {
      // stage rows colbase..colbase+15 of E_z (symmetric: gives B[k][col] columns)
      const v8bf* src = (const v8bf*)(Eb + (size_t)z * 65536 +
                                      (size_t)(colbase + srow) * 256 + schunk * 16);
      v8bf* dst = (v8bf*)(Bsh + srow * 256 + schunk * 16);
      dst[0] = src[0];
      dst[1] = src[1];
    }
    __syncthreads();
    if (active) {
      const bf16_t* brow = Bsh + q * 256;
      v8f acc = {};
      for (int kc = 0; kc < 8; kc++) {
        v8bf lo = *(const v8bf*)(brow + kc * 32 + baseB);
        v8bf hi = *(const v8bf*)(brow + kc * 32 + baseB + 16);
        v16bf B;
        for (int j = 0; j < 8; j++) { B[j] = lo[j]; B[j + 8] = hi[j]; }
        acc = __builtin_amdgcn_wmma_f32_16x16x32_bf16(false, A[kc], false, B, (short)0, acc,
                                                      false, false);
      }
      if (z == 0) {
        for (int v = 0; v < 8; v++) total[v] += ETA * acc[v];
      } else {
        int k = z - 1;
        float gk = gamma[k];
        for (int v = 0; v < 8; v++) {
          int n = rowbase + v + off8;
          float sc = ETA * gk * soft[(size_t)n * KK + k];
          total[v] -= sc * acc[v];
        }
      }
    }
  }
  if (active) {
    float thr = fabsf(thrp[0]);
    for (int v = 0; v < 8; v++) {
      int n = rowbase + v + off8;
      float h = Hn[(size_t)n * DD + col] + total[v];
      float m = fabsf(h) - thr;
      m = (m > 0.0f) ? m : 0.0f;
      out[(size_t)n * DD + col] = (h > 0.0f) ? m : ((h < 0.0f) ? -m : 0.0f);
    }
  }
}

// ---------------- launch ----------------
extern "C" void kernel_launch(void* const* d_in, const int* in_sizes, int n_in,
                              void* d_out, int out_size, void* d_ws, size_t ws_size,
                              hipStream_t stream) {
  const float* H    = (const float*)d_in[0];
  const float* Av   = (const float*)d_in[1];
  const float* soft = (const float*)d_in[2];
  const float* lnw  = (const float*)d_in[3];
  const float* lnb  = (const float*)d_in[4];
  const float* thr  = (const float*)d_in[5];
  const float* lg   = (const float*)d_in[6];
  const int* rows   = (const int*)d_in[7];
  const int* cols   = (const int*)d_in[8];
  float* out = (float*)d_out;

  char* ws = (char*)d_ws;
  size_t off = 0;
  float*  Hagg = (float*)(ws + off);  off += (size_t)NN * DD * 4;
  float*  Hn   = (float*)(ws + off);  off += (size_t)NN * DD * 4;
  bf16_t* Hnb  = (bf16_t*)(ws + off); off += (size_t)NN * DD * 2;
  bf16_t* HnT  = (bf16_t*)(ws + off); off += (size_t)DD * NN * 2;
  float*  Pt   = (float*)(ws + off);  off += (size_t)KK * NN * 4;
  float*  G    = (float*)(ws + off);  off += (size_t)11 * 65536 * 4;
  bf16_t* Eb   = (bf16_t*)(ws + off); off += (size_t)11 * 65536 * 2;
  float*  sums = (float*)(ws + off);  off += 256;
  float*  gam  = (float*)(ws + off);  off += 256;

  zero_kernel<<<2048, 256, 0, stream>>>(Hagg, (size_t)NN * DD);
  zero_kernel<<<64, 256, 0, stream>>>(G, (size_t)11 * 65536);
  zero_kernel<<<1, 32, 0, stream>>>(sums, KK);
  gamma_kernel<<<1, 32, 0, stream>>>(lg, gam);

  spmm_kernel<<<NEDGE / 8, 256, 0, stream>>>(H, Av, rows, cols, Hagg);
  ln_kernel<<<NN, 256, 0, stream>>>(Hagg, lnw, lnb, Hn, Hnb);
  transpose_kernel<<<dim3(NN / 32, DD / 32), dim3(32, 8), 0, stream>>>(
      (const unsigned short*)Hnb, (unsigned short*)HnT);
  pt_kernel<<<(NN + 255) / 256, 256, 0, stream>>>(soft, Pt, sums);
  gram_kernel<<<dim3(64, 11, 16), 32, 0, stream>>>(HnT, Pt, G);
  inv_kernel<<<11, 1024, 0, stream>>>(G, sums, Eb);
  apply_kernel<<<dim3((NN / 16 + 7) / 8, DD / 16), 256, 0, stream>>>(Hnb, Eb, Hn, soft, gam,
                                                                     thr, out);
}
